// RevealingTF_14559939134100
// MI455X (gfx1250) — matrix operand order repaired and checked by reading, not compiled, not run on hardware
//
#include <hip/hip_runtime.h>
#include <hip/hip_bf16.h>
#include <stdint.h>

// ---------------------------------------------------------------------------
// Problem constants (from reference setup_inputs)
// ---------------------------------------------------------------------------
#define N_NODES   4096
#define NODE_LEN  64
#define DIM       256
#define N_OCC     20000
#define N_VARS    2048
#define LV        8
#define N_VARNODE 8192
#define N_FUNCS   2048
#define VOCAB     50000
#define HID       1024
#define N_CLASSES 65536

typedef __attribute__((ext_vector_type(16))) _Float16 v16h;
typedef __attribute__((ext_vector_type(8)))  _Float16 v8h;
typedef __attribute__((ext_vector_type(8)))  float    v8f;

// Ordered-uint encoding of float for atomicMax-based segment max.
__device__ __forceinline__ unsigned enc_f32(float f) {
    unsigned u = __float_as_uint(f);
    return (u & 0x80000000u) ? ~u : (u | 0x80000000u);
}
// enc(-inf) = ~0xFF800000 = 0x007FFFFF  -> init value for empty segments
#define ENC_NEG_INF 0x007FFFFFu

// ---------------------------------------------------------------------------
// K0: init workspace (patch map = -1, segment-max accumulators = enc(-inf))
// ---------------------------------------------------------------------------
__global__ void k_init_ws(int* __restrict__ patch, unsigned* __restrict__ wm_enc) {
    int i = blockIdx.x * blockDim.x + threadIdx.x;
    if (i < N_NODES * NODE_LEN) patch[i]  = -1;
    if (i < N_FUNCS * DIM)      wm_enc[i] = ENC_NEG_INF;
}

// ---------------------------------------------------------------------------
// K1: scatter occurrences into the patch map.  patch[node*64+pos] = var*LV+j
// (duplicate writers race; JAX .set with duplicate indices is likewise
//  unspecified-winner, so any winner is a valid answer)
// ---------------------------------------------------------------------------
__global__ void k_scatter_occ(const int* __restrict__ occ_node,
                              const int* __restrict__ occ_begin,
                              const int* __restrict__ occ_var,
                              const int* __restrict__ pre_var_tok_lens,
                              int* __restrict__ patch) {
    int id = blockIdx.x * blockDim.x + threadIdx.x;
    if (id >= N_OCC * LV) return;
    int m = id >> 3;
    int j = id & 7;
    int var = occ_var[m];
    int pl  = pre_var_tok_lens[var];
    int beg = occ_begin[m];
    if (j < pl && (beg + pl) <= NODE_LEN) {
        patch[occ_node[m] * NODE_LEN + beg + j] = var * LV + j;
    }
}

// ---------------------------------------------------------------------------
// K2: per var-node token-mean (never materializing the 256 MB h tensor)
// fused with segment max.  One block per var-node, 256 threads = one D
// channel each.  emb_table (51 MB) stays L2-resident.
// ---------------------------------------------------------------------------
__global__ void k_gather_mean_segmax(const int*   __restrict__ var_node_index,
                                     const int*   __restrict__ seg_ids,
                                     const int*   __restrict__ node_tok_ids,
                                     const int*   __restrict__ patch,
                                     const float* __restrict__ emb_table,
                                     const float* __restrict__ pre_var_emb,
                                     unsigned*    __restrict__ wm_enc) {
    __shared__ int s_tok[NODE_LEN];
    __shared__ int s_patch[NODE_LEN];
    const int v = blockIdx.x;
    const int n = var_node_index[v];
    const int d = threadIdx.x;
    if (d < NODE_LEN) {
        s_tok[d]   = node_tok_ids[n * NODE_LEN + d];
        s_patch[d] = patch[n * NODE_LEN + d];
    }
    __syncthreads();
    float sum = 0.f;
    #pragma unroll 8
    for (int l = 0; l < NODE_LEN; ++l) {
        int p = s_patch[l];
        float val = (p >= 0) ? pre_var_emb[(size_t)p * DIM + d]
                             : emb_table[(size_t)s_tok[l] * DIM + d];
        sum += val;
    }
    float mean = sum * (1.0f / NODE_LEN);
    atomicMax(&wm_enc[(size_t)seg_ids[v] * DIM + d], enc_f32(mean));
}

// ---------------------------------------------------------------------------
// K3: decode segment max, guard non-finite (empty segments -> 0), emit f16 A
// operand for GEMM1.
// ---------------------------------------------------------------------------
__global__ void k_decode_prewm(const unsigned* __restrict__ wm_enc,
                               _Float16* __restrict__ pre_wm_h) {
    int i = blockIdx.x * blockDim.x + threadIdx.x;
    if (i >= N_FUNCS * DIM) return;
    unsigned u = wm_enc[i];
    unsigned b = (u & 0x80000000u) ? (u & 0x7FFFFFFFu) : ~u;
    float f = __uint_as_float(b);
    if ((b & 0x7F800000u) == 0x7F800000u) f = 0.0f;   // inf/nan guard
    pre_wm_h[i] = (_Float16)f;
}

// ---------------------------------------------------------------------------
// Fused transpose + f32->f16:  src[R][C] f32  ->  dst[C][R] f16.
// 32x32 LDS tile, coalesced on both sides.  R, C multiples of 32.
// Transposed weights make every WMMA B-fragment two contiguous b128 loads.
// ---------------------------------------------------------------------------
__global__ __launch_bounds__(256)
void k_transpose_f32_to_f16(const float* __restrict__ src,
                            _Float16* __restrict__ dst, int R, int C) {
    __shared__ float tile[32][33];
    const int c0 = blockIdx.x * 32;
    const int r0 = blockIdx.y * 32;
    const int tx = threadIdx.x;           // 0..31
    #pragma unroll
    for (int i = threadIdx.y; i < 32; i += 8) {
        tile[i][tx] = src[(size_t)(r0 + i) * C + (c0 + tx)];
    }
    __syncthreads();
    #pragma unroll
    for (int i = threadIdx.y; i < 32; i += 8) {
        dst[(size_t)(c0 + i) * R + (r0 + tx)] = (_Float16)tile[tx][i];
    }
}

// ---------------------------------------------------------------------------
// WMMA GEMM with pre-transposed B:
//   C[M x N] = A[M x K](f16, row-major) * BT[N x K](f16)^T + bias, f32 accum.
// Block = 256 threads = 8 waves.  Block tile: 64(M) x 256(N).
// Each wave owns a 64x32 register tile: 4 M-subtiles x 2 N-subtiles
// = 8 WMMAs per 32-wide k-step fed by 12 contiguous b128 loads
// (A frag: 2 x b128 at (M=l15, k0+8*hh) / +16;  B frag: 2 x b128 at
//  (N=l15, k0+8*hh) / +16, contiguous in K thanks to the transpose).
// blockIdx.x = M tile (fastest) so consecutive blocks reuse the same
// L2-resident BT panel; blockIdx.y = N panel.
// EP==0: +bias, store f32.   EP==1: +bias, LeakyReLU(0.3), store f16.
// ---------------------------------------------------------------------------
template <int EP, int K, int LDC>
__global__ __launch_bounds__(256)
void k_wmma_gemm_bt(const _Float16* __restrict__ A,
                    const _Float16* __restrict__ BT,
                    const float* __restrict__ bias,
                    void* __restrict__ Cout) {
    const int wave = threadIdx.x >> 5;
    const int lane = threadIdx.x & 31;
    const int hh   = lane >> 4;        // half-wave select (0/1)
    const int l15  = lane & 15;
    const int mbase = blockIdx.x * 64;
    const int nwave = blockIdx.y * 256 + wave * 32;

    v8f acc[4][2] = {};   // 4 M-subtiles x 2 N-subtiles, 8 VGPRs each

    for (int k0 = 0; k0 < K; k0 += 32) {
        // ---- 2 B fragments: e<8 -> BT[n + l15][k0 + 8*hh + e], e>=8 -> +16 ----
        union { v16h v; v8h h[2]; } bf[2];
        #pragma unroll
        for (int nt = 0; nt < 2; ++nt) {
            const _Float16* bp =
                BT + (size_t)(nwave + nt * 16 + l15) * K + k0 + 8 * hh;
            bf[nt].h[0] = *(const v8h*)bp;        // K = k0+8*hh    .. +7
            bf[nt].h[1] = *(const v8h*)(bp + 16); // K = k0+16+8*hh .. +7
        }
        // ---- 4 A fragments, each feeding both N-subtiles ----
        #pragma unroll
        for (int t = 0; t < 4; ++t) {
            const int mrow = mbase + t * 16 + l15;
            const _Float16* ap = A + (size_t)mrow * K + k0 + 8 * hh;
            union { v16h v; v8h h[2]; } af;
            af.h[0] = *(const v8h*)ap;            // K = k0+8*hh    .. +7
            af.h[1] = *(const v8h*)(ap + 16);     // K = k0+16+8*hh .. +7
            #pragma unroll
            for (int nt = 0; nt < 2; ++nt) {
                acc[t][nt] = __builtin_amdgcn_wmma_f32_16x16x32_f16(
                    false, af.v, false, bf[nt].v, (short)0, acc[t][nt],
                    false, false);
            }
        }
    }

    // ---- epilogue: D layout -> lane l15 = N, VGPR r = row 8*hh + r ----
    #pragma unroll
    for (int nt = 0; nt < 2; ++nt) {
        const int   col = nwave + nt * 16 + l15;
        const float bv  = bias[col];
        #pragma unroll
        for (int t = 0; t < 4; ++t) {
            #pragma unroll
            for (int r = 0; r < 8; ++r) {
                int row = mbase + t * 16 + 8 * hh + r;
                float y = acc[t][nt][r] + bv;
                if (EP == 1) {
                    y = (y >= 0.f) ? y : 0.3f * y;           // LeakyReLU(0.3)
                    ((_Float16*)Cout)[(size_t)row * LDC + col] = (_Float16)y;
                } else {
                    ((float*)Cout)[(size_t)row * LDC + col] = y;
                }
            }
        }
    }
}

// ---------------------------------------------------------------------------
// Launch
// ---------------------------------------------------------------------------
extern "C" void kernel_launch(void* const* d_in, const int* in_sizes, int n_in,
                              void* d_out, int out_size, void* d_ws, size_t ws_size,
                              hipStream_t stream) {
    const int*   node_tok_ids     = (const int*)  d_in[0];
    const float* pre_var_emb      = (const float*)d_in[1];
    const int*   occ_node         = (const int*)  d_in[2];
    const int*   occ_begin        = (const int*)  d_in[3];
    const int*   occ_var          = (const int*)  d_in[4];
    const int*   pre_var_tok_lens = (const int*)  d_in[5];
    const int*   var_node_index   = (const int*)  d_in[6];
    const int*   seg_ids          = (const int*)  d_in[7];
    const float* emb_table        = (const float*)d_in[8];
    const float* W1               = (const float*)d_in[9];
    const float* b1               = (const float*)d_in[10];
    const float* W2               = (const float*)d_in[11];
    const float* b2               = (const float*)d_in[12];
    float*       logits           = (float*)d_out;

    // ---- workspace carve-up (all regions fully rewritten each call) ----
    char* ws = (char*)d_ws;
    int*      patch    = (int*)ws;       ws += (size_t)4 * N_NODES * NODE_LEN;  //   1 MB
    unsigned* wm_enc   = (unsigned*)ws;  ws += (size_t)4 * N_FUNCS * DIM;       //   2 MB
    _Float16* pre_wm_h = (_Float16*)ws;  ws += (size_t)2 * N_FUNCS * DIM;       //   1 MB
    _Float16* W1t      = (_Float16*)ws;  ws += (size_t)2 * DIM * HID;           // 0.5 MB  [HID][DIM]
    _Float16* hid_h    = (_Float16*)ws;  ws += (size_t)2 * N_FUNCS * HID;       //   4 MB
    _Float16* W2t      = (_Float16*)ws;  ws += (size_t)2 * HID * N_CLASSES;     // 134 MB  [N_CLASSES][HID]

    // K0: init patch map + segment-max accumulators
    {
        int n = N_FUNCS * DIM;   // >= N_NODES*NODE_LEN
        k_init_ws<<<(n + 255) / 256, 256, 0, stream>>>(patch, wm_enc);
    }
    // Weight transpose + f16 conversion (W2t = 134 MB -> L2-resident)
    {
        dim3 blk(32, 8);
        k_transpose_f32_to_f16<<<dim3(HID / 32, DIM / 32), blk, 0, stream>>>(
            W1, W1t, DIM, HID);
        k_transpose_f32_to_f16<<<dim3(N_CLASSES / 32, HID / 32), blk, 0, stream>>>(
            W2, W2t, HID, N_CLASSES);
    }

    // K1: occurrence scatter into patch map
    k_scatter_occ<<<(N_OCC * LV + 255) / 256, 256, 0, stream>>>(
        occ_node, occ_begin, occ_var, pre_var_tok_lens, patch);

    // K2: fused gather + token-mean + segment-max
    k_gather_mean_segmax<<<N_VARNODE, 256, 0, stream>>>(
        var_node_index, seg_ids, node_tok_ids, patch, emb_table, pre_var_emb, wm_enc);

    // K3: decode + finite guard + f16 A operand
    k_decode_prewm<<<(N_FUNCS * DIM + 255) / 256, 256, 0, stream>>>(wm_enc, pre_wm_h);

    // GEMM1: [2048,256] @ [256,1024] + b1 -> LeakyReLU -> f16 hid
    {
        dim3 grid(N_FUNCS / 64, HID / 256);
        k_wmma_gemm_bt<1, DIM, HID><<<grid, 256, 0, stream>>>(pre_wm_h, W1t, b1, hid_h);
    }
    // GEMM2: [2048,1024] @ [1024,65536] + b2 -> f32 logits (275 GFLOP, WMMA)
    {
        dim3 grid(N_FUNCS / 64, N_CLASSES / 256);
        k_wmma_gemm_bt<0, HID, N_CLASSES><<<grid, 256, 0, stream>>>(hid_h, W2t, b2, logits);
    }
}
